// SparseNet_27470610825845
// MI455X (gfx1250) — matrix operand order repaired and checked by reading, not compiled, not run on hardware
//
#include <hip/hip_runtime.h>

#define DEVINL __device__ __forceinline__

typedef __attribute__((ext_vector_type(16))) __bf16 v16bf;
typedef __attribute__((ext_vector_type(8)))  __bf16 v8bf;
typedef __attribute__((ext_vector_type(8)))  float  v8f;
typedef __attribute__((ext_vector_type(4)))  float  v4f;
typedef __attribute__((ext_vector_type(4)))  unsigned int v4u;
typedef __attribute__((ext_vector_type(8)))  int    v8i;
typedef __attribute__((ext_vector_type(4)))  int    v4i;

// Problem constants (from reference setup_inputs)
constexpr int B = 2, C = 256, H = 64, W = 96;
constexpr int N = H * W;               // 6144
constexpr int NT = N / 16;             // 384 row-tiles of 16
constexpr int SEGS = 8;                // m-range split for parallelism
constexpr int MT_PER_SEG = NT / SEGS;  // 48 m-tiles per segment
constexpr int GT = 4;                  // m-tiles per TDM group (32KB)
constexpr int NGROUPS = MT_PER_SEG / GT;  // 12
constexpr int LEVELS = 5, WIN = 9;
constexpr int CAND = SEGS * 8;         // 64 merge candidates per row

// Workspace layout (bytes)
constexpr size_t SZ_MAP = (size_t)B * N * C * 2;       // bf16 map copy
constexpr size_t OFF_F1 = 0;                            // [B][N][C] rows (A side)
constexpr size_t OFF_F2 = OFF_F1 + SZ_MAP;              // B-fragment swizzled
constexpr size_t SZ_TOP = (size_t)B * N * CAND * 4;
constexpr size_t OFF_TV = OFF_F2 + SZ_MAP;
constexpr size_t OFF_TI = OFF_TV + SZ_TOP;

// ---------------- sorted top-8 insert (ascending, v[0] = current min) -------
DEVINL void topk_insert(float (&v)[8], int (&id)[8], float c, int ci) {
  bool bg[8];
#pragma unroll
  for (int j = 0; j < 8; ++j) bg[j] = c > v[j];
#pragma unroll
  for (int j = 0; j < 7; ++j) {
    v[j]  = bg[j + 1] ? v[j + 1]  : (bg[j] ? c  : v[j]);
    id[j] = bg[j + 1] ? id[j + 1] : (bg[j] ? ci : id[j]);
  }
  v[7]  = bg[7] ? c  : v[7];
  id[7] = bg[7] ? ci : id[7];
}

DEVINL v16bf cat8(v8bf lo, v8bf hi) {
  return __builtin_shufflevector(lo, hi, 0, 1, 2, 3, 4, 5, 6, 7,
                                 8, 9, 10, 11, 12, 13, 14, 15);
}

// ---------------- TDM: contiguous 32KB global -> LDS copy -------------------
// Described as a 1-D tensor: 4096 elements x 8 bytes, tile == tensor.
DEVINL void tdm_load_32k(const __bf16* gsrc, unsigned lds_off) {
  const unsigned long long ga = (unsigned long long)(size_t)gsrc;
  v4u g0;
  g0[0] = 1u;                                   // count=1, user mode
  g0[1] = lds_off;                              // lds_addr (bytes)
  g0[2] = (unsigned)(ga & 0xFFFFFFFFu);         // global_addr[31:0]
  g0[3] = (unsigned)((ga >> 32) & 0x01FFFFFFu)  // global_addr[56:32]
          | 0x80000000u;                        // type=2 ("image")
  v8i g1;
  g1[0] = 0x00030000;   // workgroup_mask=0, data_size=3 (8B), no flags
  g1[1] = 0x10000000;   // tensor_dim0 = 4096 (low 16 bits in [31:16])
  g1[2] = 0x00010000;   // tensor_dim0 hi=0, tensor_dim1 = 1
  g1[3] = 0x10000000;   // tensor_dim1 hi=0, tile_dim0 = 4096
  g1[4] = 0;            // tile_dim1=0 (unused), tile_dim2=0
  g1[5] = 0x00001000;   // tensor_dim0_stride = 4096 (low 32)
  g1[6] = 0;            // stride hi, tensor_dim1_stride lo
  g1[7] = 0;
  const v4i z4 = {0, 0, 0, 0};
#if __clang_major__ >= 23
  const v8i z8 = {0, 0, 0, 0, 0, 0, 0, 0};
  __builtin_amdgcn_tensor_load_to_lds(g0, g1, z4, z4, z8, 0);
#else
  __builtin_amdgcn_tensor_load_to_lds(g0, g1, z4, z4, 0);
#endif
}

// ---------------- kernel 1: cast f32 [B,C,N] -> bf16 operand layouts --------
// z==0: fmap1 -> row-major [B][N][C]              (A-fragment source)
// z==1: fmap2 -> [B][mtile][kchunk][lane][16]     (exact WMMA B-fragment order)
__global__ __launch_bounds__(256) void sn_cast_transpose(
    const float* __restrict__ s1, const float* __restrict__ s2,
    __bf16* __restrict__ d1, __bf16* __restrict__ d2) {
  const int nb = blockIdx.x;      // N/32 tiles of 32 pixels (= 2 m-tiles)
  const int b  = blockIdx.y;
  const float* src = blockIdx.z ? s2 : s1;
  __bf16*      dst = blockIdx.z ? d2 : d1;
  src += (size_t)b * C * N;
  dst += (size_t)b * N * C;
  __shared__ __bf16 tile[32][C + 8];
  const int n0 = nb * 32;
  const int nl = threadIdx.x & 31;
  const int cg = threadIdx.x >> 5;  // 0..7
  for (int cc = 0; cc < C; cc += 8) {
    const int c = cc + cg;
    tile[nl][c] = (__bf16)src[(size_t)c * N + n0 + nl];  // coalesced over n
  }
  __syncthreads();
  if (blockIdx.z == 0) {
    const int row = threadIdx.x >> 3;
    const int sub = threadIdx.x & 7;
    __bf16* drow = dst + (size_t)(n0 + row) * C + sub * 32;
    const __bf16* srow = &tile[row][sub * 32];
#pragma unroll
    for (int q = 0; q < 4; ++q)
      *(v8bf*)(drow + q * 8) = *(const v8bf*)(srow + q * 8);
  } else {
    const int tl = threadIdx.x >> 7;  // which of the 2 m-tiles
#pragma unroll
    for (int s2i = 0; s2i < 2; ++s2i) {
      const int s  = ((threadIdx.x & 127) << 1) | s2i;  // slot 0..255
      const int kc = s >> 5;
      const int ln = s & 31;
      const int r  = ln & 15, hh = ln >> 4;
      const __bf16* sp = &tile[tl * 16 + r][kc * 32 + 8 * hh];
      __bf16* dp = dst + (size_t)(nb * 2 + tl) * 4096 + kc * 512 + ln * 16;
      *(v8bf*)(dp)     = *(const v8bf*)(sp);
      *(v8bf*)(dp + 8) = *(const v8bf*)(sp + 16);
    }
  }
}

// ---------------- kernel 2: zero output ------------------------------------
__global__ __launch_bounds__(256) void sn_zero(float* __restrict__ p, int n) {
  const int i = blockIdx.x * blockDim.x + threadIdx.x;
  if (i < n) p[i] = 0.0f;
}

// ---------------- kernel 3: TDM-fed bf16 WMMA GEMM + fused top-8 ------------
__global__ __launch_bounds__(256) void sn_gemm_topk(
    const __bf16* __restrict__ f1bf, const __bf16* __restrict__ f2sw,
    float* __restrict__ topv, int* __restrict__ topi) {
  const int lane = threadIdx.x & 31;
  const int wave = threadIdx.x >> 5;
  const int bx   = blockIdx.x;
  const int seg  = bx & (SEGS - 1);
  const int rest = bx >> 3;
  const int nt8  = rest % (NT / 8);
  const int b    = rest / (NT / 8);
  const int n0   = (nt8 * 8 + wave) * 16;
  const int r    = lane & 15;       // row / col within tile
  const int h    = lane >> 4;       // K-half selector

  __shared__ __align__(128) __bf16 bbuf[2][GT * 4096];  // 2 x 32KB double buffer
  __shared__ float cs[8][16 * 20];                      // per-wave spill tiles

  const __bf16* arow = f1bf + (size_t)b * N * C + (size_t)(n0 + r) * C;
  const __bf16* f2b  = f2sw + (size_t)b * N * C;

  // A tile (16 rows x C) register-resident as 8 k-chunk fragments.
  v16bf afrag[8];
#pragma unroll
  for (int kc = 0; kc < 8; ++kc) {
    v8bf lo = *(const v8bf*)(arow + kc * 32 + 8 * h);
    v8bf hi = *(const v8bf*)(arow + kc * 32 + 16 + 8 * h);
    afrag[kc] = cat8(lo, hi);
  }

  float tv[8]; int ti[8];
#pragma unroll
  for (int j = 0; j < 8; ++j) { tv[j] = -3.4028235e38f; ti[j] = 0; }

  // make the double buffer well-defined before TDM takes over
  {
    float* zb = (float*)&bbuf[0][0];
    for (int i = threadIdx.x; i < 2 * GT * 4096 / 2; i += 256) zb[i] = 0.0f;
  }
  __syncthreads();

  const int mt0 = seg * MT_PER_SEG;
  const __bf16* gseg = f2b + (size_t)mt0 * 4096;
  if (wave == 0) {
    tdm_load_32k(gseg,                 (unsigned)(size_t)&bbuf[0][0]);
    tdm_load_32k(gseg + GT * 4096,     (unsigned)(size_t)&bbuf[1][0]);
  }

  float* lds = cs[wave];
  for (int g = 0; g < NGROUPS; ++g) {
    if (wave == 0) {
      if (g + 1 < NGROUPS) __builtin_amdgcn_s_wait_tensorcnt(1);
      else                 __builtin_amdgcn_s_wait_tensorcnt(0);
    }
    __syncthreads();  // publish bbuf[g&1]
    const __bf16* bgrp = &bbuf[g & 1][0];
#pragma unroll 1
    for (int tt = 0; tt < GT; ++tt) {
      const int m0 = (mt0 + g * GT + tt) * 16;
      const __bf16* bb = bgrp + tt * 4096 + lane * 16;
      v16bf bfrag[8];
#pragma unroll
      for (int kc = 0; kc < 8; ++kc) {
        v8bf blo = *(const v8bf*)(bb + kc * 512);
        v8bf bhi = *(const v8bf*)(bb + kc * 512 + 8);
        bfrag[kc] = cat8(blo, bhi);
      }
      v8f acc = {};
#pragma unroll
      for (int kc = 0; kc < 8; ++kc)
        acc = __builtin_amdgcn_wmma_f32_16x16x32_bf16(
            false, afrag[kc], false, bfrag[kc], (short)0, acc, false, false);

      // C layout: lane holds col r, rows v + 8h -> transpose via per-wave LDS.
      asm volatile("" ::: "memory");
#pragma unroll
      for (int v = 0; v < 8; ++v) lds[(v + 8 * h) * 20 + r] = acc[v];
      asm volatile("" ::: "memory");
      const float* lrow = lds + r * 20 + 8 * h;  // lane owns row r, 8 cols
      v4f c0 = *(const v4f*)(lrow);
      v4f c1 = *(const v4f*)(lrow + 4);
      asm volatile("" ::: "memory");
      const float mx =
          fmaxf(fmaxf(fmaxf(c0[0], c0[1]), fmaxf(c0[2], c0[3])),
                fmaxf(fmaxf(c1[0], c1[1]), fmaxf(c1[2], c1[3])));
      if (mx > tv[0]) {  // group prefilter
        const int cb = m0 + 8 * h;
#pragma unroll
        for (int j = 0; j < 8; ++j) {
          const float c = (j < 4) ? c0[j] : c1[j - 4];
          if (c > tv[0]) topk_insert(tv, ti, c, cb + j);
        }
      }
    }
    __syncthreads();  // all waves done reading bbuf[g&1]
    if (wave == 0 && g + 2 < NGROUPS)
      tdm_load_32k(gseg + (size_t)(g + 2) * GT * 4096,
                   (unsigned)(size_t)&bbuf[g & 1][0]);
  }

  // merge high-half partial lists into low-half lanes (same-wave DS ordering)
  asm volatile("" ::: "memory");
  if (h == 1) {
#pragma unroll
    for (int j = 0; j < 8; ++j) {
      lds[r * 20 + j]     = tv[j];
      lds[r * 20 + 8 + j] = __int_as_float(ti[j]);
    }
  }
  asm volatile("" ::: "memory");
  if (h == 0) {
#pragma unroll
    for (int j = 0; j < 8; ++j) {
      const float c  = lds[r * 20 + j];
      const int   ci = __float_as_int(lds[r * 20 + 8 + j]);
      if (c > tv[0]) topk_insert(tv, ti, c, ci);
    }
    const size_t base = ((size_t)(b * N + n0 + r) * SEGS + seg) * 8;
#pragma unroll
    for (int j = 0; j < 8; ++j) { topv[base + j] = tv[j]; topi[base + j] = ti[j]; }
  }
}

// ---------------- kernel 4: merge segments + bilinear pyramid scatter -------
__global__ __launch_bounds__(256) void sn_merge_scatter(
    const float* __restrict__ topv, const int* __restrict__ topi,
    const float* __restrict__ flow, float* __restrict__ out) {
  const int t = blockIdx.x * blockDim.x + threadIdx.x;
  if (t >= B * N) return;
  const int b = t / N, n = t % N;

  float tv[8]; int ti[8];
#pragma unroll
  for (int j = 0; j < 8; ++j) { tv[j] = -3.4028235e38f; ti[j] = 0; }
  const float* vs = topv + (size_t)t * CAND;
  const int*   is = topi + (size_t)t * CAND;
  for (int j = 0; j < CAND; ++j) {
    const float c = vs[j]; const int ci = is[j];
    if (c > tv[0]) topk_insert(tv, ti, c, ci);
  }

  const int y1 = n / W, x1 = n % W;
  const float fly = flow[((size_t)b * 2 + 1) * N + n];  // flow[:, ::-1] -> y
  const float flx = flow[((size_t)b * 2 + 0) * N + n];  //               -> x
  float* outb = out + (size_t)b * (LEVELS * WIN * WIN) * N + n;

#pragma unroll
  for (int k = 0; k < 8; ++k) {
    const float corr = tv[k] * 0.0625f;                 // / sqrt(C)
    const int m = ti[k];
    const int y2 = m / W, x2 = m % W;
    const float dy = (float)(y2 - y1) - fly;
    const float dx = (float)(x2 - x1) - flx;
    float scale = 1.0f;
    for (int lvl = 0; lvl < LEVELS; ++lvl) {
      const float cy = dy * scale, cx = dx * scale;
      const float fy = floorf(cy), fx = floorf(cx);
#pragma unroll
      for (int cnr = 0; cnr < 4; ++cnr) {
        const float ccy = fy + (float)(cnr >> 1);
        const float ccx = fx + (float)(cnr & 1);
        if (fabsf(ccy) <= 4.0f && fabsf(ccx) <= 4.0f) {
          const float w = (1.0f - fabsf(cy - ccy)) * (1.0f - fabsf(cx - ccx));
          const int ch = lvl * 81 + ((int)ccy + 4) * 9 + ((int)ccx + 4);
          atomicAdd(outb + (size_t)ch * N, w * corr);
        }
      }
      scale *= 0.5f;
    }
  }
}

// ---------------- host launcher --------------------------------------------
extern "C" void kernel_launch(void* const* d_in, const int* in_sizes, int n_in,
                              void* d_out, int out_size, void* d_ws, size_t ws_size,
                              hipStream_t stream) {
  (void)in_sizes; (void)n_in; (void)ws_size;
  const float* fmap1 = (const float*)d_in[0];
  const float* fmap2 = (const float*)d_in[1];
  const float* flow  = (const float*)d_in[2];
  float* out = (float*)d_out;
  char* ws = (char*)d_ws;
  __bf16* f1bf = (__bf16*)(ws + OFF_F1);
  __bf16* f2sw = (__bf16*)(ws + OFF_F2);
  float*  topv = (float*)(ws + OFF_TV);
  int*    topi = (int*)(ws + OFF_TI);

  dim3 tgrid(N / 32, B, 2);
  sn_cast_transpose<<<tgrid, 256, 0, stream>>>(fmap1, fmap2, f1bf, f2sw);
  sn_zero<<<(out_size + 255) / 256, 256, 0, stream>>>(out, out_size);
  sn_gemm_topk<<<B * (NT / 8) * SEGS, 256, 0, stream>>>(f1bf, f2sw, topv, topi);
  sn_merge_scatter<<<(B * N + 255) / 256, 256, 0, stream>>>(topv, topi, flow, out);
}
